// WindowAttention_283467842161
// MI455X (gfx1250) — matrix operand order repaired and checked by reading, not compiled, not run on hardware
//
#include <hip/hip_runtime.h>
#include <hip/hip_bf16.h>

typedef __bf16 v16bf __attribute__((ext_vector_type(16)));
typedef __bf16 v8bf  __attribute__((ext_vector_type(8)));
typedef float  v8f   __attribute__((ext_vector_type(8)));

union V16 { v16bf v; v8bf h[2]; };

#define V8F_ZERO {0.f,0.f,0.f,0.f,0.f,0.f,0.f,0.f}

static __device__ __forceinline__ v8f wmma_bf16(const V16& a, const V16& b, v8f c) {
  return __builtin_amdgcn_wmma_f32_16x16x32_bf16(false, a.v, false, b.v, (short)0, c, false, false);
}

// ---------------------------------------------------------------------------
// k_prep: weights fp32->bf16, materialize bias[h][m][n] = rpb[rel_index[m][n]][h]
// ---------------------------------------------------------------------------
__global__ void k_prep(const float* __restrict__ qkv_w, const float* __restrict__ proj_w,
                       const float* __restrict__ rpb, const int* __restrict__ reli,
                       __bf16* __restrict__ wqkv, __bf16* __restrict__ wproj,
                       float* __restrict__ bias) {
  int i = blockIdx.x * 256 + threadIdx.x;
  const int NQ = 1536 * 512, NP = 512 * 512, NB = 16 * 4096;
  if (i < NQ) { wqkv[i] = (__bf16)qkv_w[i]; return; }
  i -= NQ;
  if (i < NP) { wproj[i] = (__bf16)proj_w[i]; return; }
  i -= NP;
  if (i < NB) {
    int h = i >> 12, mn = i & 4095;
    bias[i] = rpb[reli[mn] * 16 + h];
  }
}

// ---------------------------------------------------------------------------
// k_qkv: qkv = x @ qkv_w^T + qkv_b ; scatter to q (scaled), k, v^T  (bf16)
// 64-row M block, LDS-staged bf16 A with xor-swizzled 16B chunks.
// 4 waves; wave w owns N-tiles [w*24, w*24+24) in 6 groups of 4.
// ---------------------------------------------------------------------------
__global__ __launch_bounds__(128) void k_qkv(
    const float* __restrict__ x, const __bf16* __restrict__ wqkv,
    const float* __restrict__ qkv_b,
    __bf16* __restrict__ qbuf, __bf16* __restrict__ kbuf, __bf16* __restrict__ vtbuf) {
  __shared__ __align__(16) __bf16 As[64 * 512];  // 64 KB
  const int tid = threadIdx.x;
  const int m0 = blockIdx.x * 64;

  for (int i = tid; i < 64 * 64; i += 128) {     // 16B chunks
    int row = i >> 6, ch = i & 63;
    const float4* s = (const float4*)(x + (size_t)(m0 + row) * 512 + ch * 8);
    float4 f0 = s[0], f1 = s[1];
    v8bf hh;
    hh[0] = (__bf16)f0.x; hh[1] = (__bf16)f0.y; hh[2] = (__bf16)f0.z; hh[3] = (__bf16)f0.w;
    hh[4] = (__bf16)f1.x; hh[5] = (__bf16)f1.y; hh[6] = (__bf16)f1.z; hh[7] = (__bf16)f1.w;
    *(v8bf*)((char*)As + row * 1024 + ((ch ^ (row & 15)) << 4)) = hh;
  }
  __syncthreads();

  const int wave = tid >> 5, lane = tid & 31, half = lane >> 4, ln = lane & 15;
  const float QSCALE = 0.17677669529663687f;  // 32^-0.5

  for (int g = 0; g < 6; ++g) {
    const int nb = wave * 24 + g * 4;
    v8f c[4][4];
#pragma unroll
    for (int mt = 0; mt < 4; ++mt)
#pragma unroll
      for (int t = 0; t < 4; ++t) c[mt][t] = (v8f)V8F_ZERO;

    for (int kc = 0; kc < 16; ++kc) {
      V16 a[4];
#pragma unroll
      for (int mt = 0; mt < 4; ++mt) {
        int row = mt * 16 + ln;
        int c1 = (kc * 4 + half)     ^ (row & 15);
        int c2 = (kc * 4 + half + 2) ^ (row & 15);
        a[mt].h[0] = *(const v8bf*)((const char*)As + row * 1024 + (c1 << 4));
        a[mt].h[1] = *(const v8bf*)((const char*)As + row * 1024 + (c2 << 4));
      }
#pragma unroll
      for (int t = 0; t < 4; ++t) {
        int n = (nb + t) * 16 + ln;
        const v8bf* bp = (const v8bf*)(wqkv + (size_t)n * 512 + kc * 32 + half * 8);
        V16 b; b.h[0] = bp[0]; b.h[1] = bp[2];
#pragma unroll
        for (int mt = 0; mt < 4; ++mt) c[mt][t] = wmma_bf16(a[mt], b, c[mt][t]);
      }
    }

#pragma unroll
    for (int t = 0; t < 4; ++t) {
      int ncol = (nb + t) * 16;
      int sec = ncol >> 9;         // 0=q 1=k 2=v
      int within = ncol & 511;
      int hd = within >> 5;        // head
      int dbase = within & 31;     // 0 or 16
      float bv = qkv_b[ncol + ln];
#pragma unroll
      for (int mt = 0; mt < 4; ++mt) {
#pragma unroll
        for (int r = 0; r < 8; ++r) {
          int m = m0 + mt * 16 + half * 8 + r;
          int bwin = m >> 6, tok = m & 63;
          float val = c[mt][t][r] + bv;
          if (sec == 0) {
            qbuf[(((size_t)bwin * 16 + hd) * 64 + tok) * 32 + dbase + ln] = (__bf16)(val * QSCALE);
          } else if (sec == 1) {
            kbuf[(((size_t)bwin * 16 + hd) * 64 + tok) * 32 + dbase + ln] = (__bf16)val;
          } else {
            vtbuf[(((size_t)bwin * 16 + hd) * 32 + dbase + ln) * 64 + tok] = (__bf16)val;
          }
        }
      }
    }
  }
}

// ---------------------------------------------------------------------------
// k_attn: one block per (window,head); wave w handles token rows [16w,16w+16)
// S = Q K^T (WMMA) -> +bias+mask -> softmax (half-wave shfl reductions)
// -> P bf16 to LDS -> O = P V (WMMA) -> obuf[b][t][h*32+d]
// ---------------------------------------------------------------------------
__global__ __launch_bounds__(128) void k_attn(
    const __bf16* __restrict__ qbuf, const __bf16* __restrict__ kbuf,
    const __bf16* __restrict__ vtbuf, const float* __restrict__ bias,
    const float* __restrict__ mask, __bf16* __restrict__ obuf) {
  __shared__ __align__(16) __bf16 P[4 * 16 * 72];  // pitch 72 avoids bank conflicts
  const int tid = threadIdx.x;
  const int wave = tid >> 5, lane = tid & 31, half = lane >> 4, ln = lane & 15;
  const int b = blockIdx.x >> 4, h = blockIdx.x & 15, nw = b & 63;
  const __bf16* qp  = qbuf + ((size_t)b * 16 + h) * 64 * 32;
  const __bf16* kp  = kbuf + ((size_t)b * 16 + h) * 64 * 32;
  const __bf16* vtp = vtbuf + ((size_t)b * 16 + h) * 32 * 64;

  V16 aq;
  {
    int row = wave * 16 + ln;
    aq.h[0] = *(const v8bf*)(qp + row * 32 + half * 8);
    aq.h[1] = *(const v8bf*)(qp + row * 32 + half * 8 + 16);
  }
  v8f c[4];
#pragma unroll
  for (int nt = 0; nt < 4; ++nt) {
    int n = nt * 16 + ln;
    const v8bf* bp = (const v8bf*)(kp + n * 32 + half * 8);
    V16 bk; bk.h[0] = bp[0]; bk.h[1] = bp[2];
    v8f z = V8F_ZERO;
    c[nt] = wmma_bf16(aq, bk, z);
  }

  const float* bs = bias + h * 4096;
  const float* mk = mask + nw * 4096;
#pragma unroll
  for (int nt = 0; nt < 4; ++nt)
#pragma unroll
    for (int r = 0; r < 8; ++r) {
      int m = wave * 16 + half * 8 + r;
      int n = nt * 16 + ln;
      c[nt][r] += bs[m * 64 + n] + mk[m * 64 + n];
    }

  __bf16* Pw = &P[wave * 16 * 72];
#pragma unroll
  for (int r = 0; r < 8; ++r) {
    float mx = fmaxf(fmaxf(c[0][r], c[1][r]), fmaxf(c[2][r], c[3][r]));
    mx = fmaxf(mx, __shfl_xor(mx, 1));
    mx = fmaxf(mx, __shfl_xor(mx, 2));
    mx = fmaxf(mx, __shfl_xor(mx, 4));
    mx = fmaxf(mx, __shfl_xor(mx, 8));   // stays within 16-lane half
    float e0 = __expf(c[0][r] - mx), e1 = __expf(c[1][r] - mx);
    float e2 = __expf(c[2][r] - mx), e3 = __expf(c[3][r] - mx);
    float sm = e0 + e1 + e2 + e3;
    sm += __shfl_xor(sm, 1); sm += __shfl_xor(sm, 2);
    sm += __shfl_xor(sm, 4); sm += __shfl_xor(sm, 8);
    float inv = 1.0f / sm;
    int rowl = half * 8 + r;
    Pw[rowl * 72 +  0 + ln] = (__bf16)(e0 * inv);
    Pw[rowl * 72 + 16 + ln] = (__bf16)(e1 * inv);
    Pw[rowl * 72 + 32 + ln] = (__bf16)(e2 * inv);
    Pw[rowl * 72 + 48 + ln] = (__bf16)(e3 * inv);
  }
  // own-wave LDS ops are in order; each wave only reads rows it wrote.

  V16 ap[2];
#pragma unroll
  for (int kc = 0; kc < 2; ++kc) {
    int k0 = kc * 32 + half * 8;
    ap[kc].h[0] = *(const v8bf*)((const char*)Pw + ln * 144 + k0 * 2);
    ap[kc].h[1] = *(const v8bf*)((const char*)Pw + ln * 144 + (k0 + 16) * 2);
  }
  v8f o[2] = {V8F_ZERO, V8F_ZERO};
#pragma unroll
  for (int nt2 = 0; nt2 < 2; ++nt2) {
#pragma unroll
    for (int kc = 0; kc < 2; ++kc) {
      int d = nt2 * 16 + ln;
      const v8bf* bp = (const v8bf*)(vtp + d * 64 + kc * 32 + half * 8);
      V16 bv; bv.h[0] = bp[0]; bv.h[1] = bp[2];
      o[nt2] = wmma_bf16(ap[kc], bv, o[nt2]);
    }
  }
#pragma unroll
  for (int nt2 = 0; nt2 < 2; ++nt2)
#pragma unroll
    for (int r = 0; r < 8; ++r) {
      int tok = wave * 16 + half * 8 + r;
      int d = nt2 * 16 + ln;
      obuf[((size_t)b * 64 + tok) * 512 + h * 32 + d] = (__bf16)o[nt2][r];
    }
}

// ---------------------------------------------------------------------------
// k_proj: out = O @ proj_w^T + proj_b  (fp32 out). Same tiling as k_qkv.
// Wave w owns N-tiles [w*8, w*8+8) in 2 groups of 4.
// ---------------------------------------------------------------------------
__global__ __launch_bounds__(128) void k_proj(
    const __bf16* __restrict__ obuf, const __bf16* __restrict__ wproj,
    const float* __restrict__ proj_b, float* __restrict__ out) {
  __shared__ __align__(16) __bf16 As[64 * 512];
  const int tid = threadIdx.x;
  const int m0 = blockIdx.x * 64;
  for (int i = tid; i < 64 * 64; i += 128) {
    int row = i >> 6, ch = i & 63;
    v8bf v = *(const v8bf*)(obuf + (size_t)(m0 + row) * 512 + ch * 8);
    *(v8bf*)((char*)As + row * 1024 + ((ch ^ (row & 15)) << 4)) = v;
  }
  __syncthreads();

  const int wave = tid >> 5, lane = tid & 31, half = lane >> 4, ln = lane & 15;
  for (int g = 0; g < 2; ++g) {
    const int nb = wave * 8 + g * 4;
    v8f c[4][4];
#pragma unroll
    for (int mt = 0; mt < 4; ++mt)
#pragma unroll
      for (int t = 0; t < 4; ++t) c[mt][t] = (v8f)V8F_ZERO;

    for (int kc = 0; kc < 16; ++kc) {
      V16 a[4];
#pragma unroll
      for (int mt = 0; mt < 4; ++mt) {
        int row = mt * 16 + ln;
        int c1 = (kc * 4 + half)     ^ (row & 15);
        int c2 = (kc * 4 + half + 2) ^ (row & 15);
        a[mt].h[0] = *(const v8bf*)((const char*)As + row * 1024 + (c1 << 4));
        a[mt].h[1] = *(const v8bf*)((const char*)As + row * 1024 + (c2 << 4));
      }
#pragma unroll
      for (int t = 0; t < 4; ++t) {
        int n = (nb + t) * 16 + ln;
        const v8bf* bp = (const v8bf*)(wproj + (size_t)n * 512 + kc * 32 + half * 8);
        V16 b; b.h[0] = bp[0]; b.h[1] = bp[2];
#pragma unroll
        for (int mt = 0; mt < 4; ++mt) c[mt][t] = wmma_bf16(a[mt], b, c[mt][t]);
      }
    }
#pragma unroll
    for (int t = 0; t < 4; ++t) {
      int n = (nb + t) * 16 + ln;
      float pb = proj_b[n];
#pragma unroll
      for (int mt = 0; mt < 4; ++mt)
#pragma unroll
        for (int r = 0; r < 8; ++r) {
          int m = m0 + mt * 16 + half * 8 + r;
          out[(size_t)m * 512 + n] = c[mt][t][r] + pb;
        }
    }
  }
}

// ---------------------------------------------------------------------------
extern "C" void kernel_launch(void* const* d_in, const int* in_sizes, int n_in,
                              void* d_out, int out_size, void* d_ws, size_t ws_size,
                              hipStream_t stream) {
  const float* x      = (const float*)d_in[0];
  const float* mask   = (const float*)d_in[1];
  const float* qkv_w  = (const float*)d_in[2];
  const float* qkv_b  = (const float*)d_in[3];
  const float* proj_w = (const float*)d_in[4];
  const float* proj_b = (const float*)d_in[5];
  const float* rpb    = (const float*)d_in[6];
  const int*   reli   = (const int*)d_in[7];
  float* out = (float*)d_out;

  char* ws = (char*)d_ws;
  const size_t SZQ = (size_t)1024 * 16 * 64 * 32 * 2;  // 64 MiB per buffer
  __bf16* qbuf  = (__bf16*)(ws);
  __bf16* kbuf  = (__bf16*)(ws + SZQ);
  __bf16* vtbuf = (__bf16*)(ws + 2 * SZQ);
  __bf16* obuf  = (__bf16*)(ws + 3 * SZQ);
  __bf16* wqkv  = (__bf16*)(ws + 4 * SZQ);
  __bf16* wproj = (__bf16*)(ws + 4 * SZQ + (size_t)1536 * 512 * 2);
  float*  bias  = (float*) (ws + 4 * SZQ + (size_t)1536 * 512 * 2 + (size_t)512 * 512 * 2);

  const int totalPrep = 1536 * 512 + 512 * 512 + 16 * 4096;
  k_prep<<<(totalPrep + 255) / 256, 256, 0, stream>>>(qkv_w, proj_w, rpb, reli, wqkv, wproj, bias);
  k_qkv<<<1024, 128, 0, stream>>>(x, wqkv, qkv_b, qbuf, kbuf, vtbuf);
  k_attn<<<16384, 128, 0, stream>>>(qbuf, kbuf, vtbuf, bias, mask, obuf);
  k_proj<<<1024, 128, 0, stream>>>(obuf, wproj, proj_b, out);
}
